// HMLSTMCell_41540923687172
// MI455X (gfx1250) — compile-verified, matrix operand order checked by reading, not run on hardware
//
#include <hip/hip_runtime.h>
#include <hip/hip_bf16.h>
#include <math.h>

#define H     1024
#define Bn    2048
#define KTOT  3072
#define KT    (KTOT / 32)   // 96 k-tiles of 32
#define MT    (H / 16)      // 64 m-tiles per gate
#define NT    (Bn / 16)     // 128 n-tiles

typedef __bf16 bf16_t;
typedef bf16_t bf16x16 __attribute__((ext_vector_type(16)));
typedef float  f32x8   __attribute__((ext_vector_type(8)));

// ---------------------------------------------------------------------------
// Fragment-major bf16 packing, matching the CDNA5 WMMA 16-bit A/B layouts:
// lane 0-15  : row/col = lane,      K elements {0..7, 16..23}
// lane 16-31 : row/col = lane-16,   K elements {8..15, 24..31}
// Fragment = 32 lanes x 16 bf16 (1 KB); a wave loads it with 2x b128/lane.
// ---------------------------------------------------------------------------

__device__ __forceinline__ int frag_kk(int e, int half) {
    return (e & 7) + ((e >> 3) << 4) + (half << 3);
}

// Pack [W_hh | U_11 | U_21] rows 0..4095 (4 gates x 1024) into bf16 fragments.
__global__ __launch_bounds__(256) void pack_A(const float* __restrict__ Whh,
                                              const float* __restrict__ U11,
                                              const float* __restrict__ U21,
                                              bf16_t* __restrict__ Apack) {
    int t = blockIdx.x * blockDim.x + threadIdx.x;
    int lane = t & 31; t >>= 5;
    int kt = t % KT;  t /= KT;
    int mt = t % MT;
    int gate = t / MT;                       // 0..3
    int r    = gate * H + mt * 16 + (lane & 15);
    int half = lane >> 4;

    bf16x16 v;
#pragma unroll
    for (int e = 0; e < 16; ++e) {
        int k   = kt * 32 + frag_kk(e, half);    // global K in [0,3072)
        int src = k >> 10;
        int kin = k & 1023;
        const float* __restrict__ W = (src == 0) ? Whh : (src == 1) ? U11 : U21;
        v[e] = (bf16_t)W[(size_t)r * H + kin];
    }
    bf16_t* dst = Apack + ((size_t)((gate * MT + mt) * KT + kt) * 32 + lane) * 16;
    *(bf16x16*)dst = v;
}

// Pack [h_bottom ; z*h ; z*h_top] (3072 x 2048) into bf16 B fragments.
__global__ __launch_bounds__(256) void pack_B(const float* __restrict__ hb,
                                              const float* __restrict__ h,
                                              const float* __restrict__ ht,
                                              const float* __restrict__ z,
                                              bf16_t* __restrict__ Bpack) {
    int t = blockIdx.x * blockDim.x + threadIdx.x;
    int lane = t & 31; t >>= 5;
    int kt = t % KT;
    int nt = t / KT;                         // 0..127
    int n    = nt * 16 + (lane & 15);
    int half = lane >> 4;
    float zn = z[n];

    bf16x16 v;
#pragma unroll
    for (int e = 0; e < 16; ++e) {
        int k   = kt * 32 + frag_kk(e, half);
        int src = k >> 10;
        int kin = k & 1023;
        float val;
        if (src == 0)      val = hb[(size_t)kin * Bn + n];
        else if (src == 1) val = zn * h[(size_t)kin * Bn + n];
        else               val = zn * ht[(size_t)kin * Bn + n];
        v[e] = (bf16_t)val;
    }
    bf16_t* dst = Bpack + ((size_t)(nt * KT + kt) * 32 + lane) * 16;
    *(bf16x16*)dst = v;
}

__device__ __forceinline__ float sigmoidf_(float x) {
    return 1.0f / (1.0f + __expf(-x));
}

// ---------------------------------------------------------------------------
// Fused GEMM + HM-LSTM epilogue.
// Workgroup: 256 threads = 8 waves (2 M-waves x 4 N-waves) -> 32 x 256 tile.
// Each wave: one 16-row m-tile across ALL 4 gates x FOUR 16-col n-tiles
//            = 16 accumulators (128 f32 VGPRs); per k-step: 8 fragment loads
//            feed 16 v_wmma (2.0 WMMA/fragment -> halves L2 traffic vs 1.33).
// ---------------------------------------------------------------------------
__global__ __launch_bounds__(256) void hmlstm_gemm(
        const bf16_t* __restrict__ Apack, const bf16_t* __restrict__ Bpack,
        const float* __restrict__ bias,   const float* __restrict__ c,
        const float* __restrict__ h,      const float* __restrict__ z,
        const float* __restrict__ zb,     float* __restrict__ out) {
    int lane  = threadIdx.x & 31;
    int wave  = threadIdx.x >> 5;
    int waveN = wave & 3;
    int waveM = wave >> 2;
    int nBlock = blockIdx.x & 7;           // 8 n-blocks of 256
    int mBlock = blockIdx.x >> 3;          // 32 m-blocks of 32
    int mTile  = mBlock * 2 + waveM;       // 0..63
    int nt0    = nBlock * 16 + waveN * 4;  // 0..124

    f32x8 acc[4][4];
#pragma unroll
    for (int g = 0; g < 4; ++g)
#pragma unroll
        for (int j = 0; j < 4; ++j) acc[g][j] = (f32x8)0.0f;

    const size_t kStep = (size_t)32 * 16;        // 1 KB fragment per k-tile
    const size_t laneOff = (size_t)lane * 16;

    // Per-fragment base pointers (k-tile 0); the k loop adds a uniform offset.
    const bf16_t* aPtr[4];
#pragma unroll
    for (int g = 0; g < 4; ++g)
        aPtr[g] = Apack + (size_t)((g * MT + mTile) * KT) * 512 + laneOff;
    const bf16_t* bPtr[4];
#pragma unroll
    for (int j = 0; j < 4; ++j)
        bPtr[j] = Bpack + (size_t)((nt0 + j) * KT) * 512 + laneOff;

#pragma unroll 2
    for (int kt = 0; kt < KT; ++kt) {
        size_t ko = (size_t)kt * kStep;
        bf16x16 a[4], b[4];
#pragma unroll
        for (int g = 0; g < 4; ++g) a[g] = *(const bf16x16*)(aPtr[g] + ko);
#pragma unroll
        for (int j = 0; j < 4; ++j) b[j] = *(const bf16x16*)(bPtr[j] + ko);
        // Pull the next k-tile toward L2/L0 (global_prefetch_b8).
        __builtin_prefetch(aPtr[0] + ko + kStep, 0, 1);
        __builtin_prefetch(bPtr[0] + ko + kStep, 0, 1);
#pragma unroll
        for (int g = 0; g < 4; ++g)
#pragma unroll
            for (int j = 0; j < 4; ++j)
                acc[g][j] = __builtin_amdgcn_wmma_f32_16x16x32_bf16(
                    false, a[g], false, b[j], (short)0, acc[g][j], false, false);
    }

    // ---- epilogue: C/D layout: VGPR v, lanes 0-15 -> M=v, lanes 16-31 -> M=v+8
    int half  = lane >> 4;
    int nl    = lane & 15;
    int mBase = mTile * 16 + half * 8;

    float bF[8], bI[8], bO[8], bG[8];
#pragma unroll
    for (int v = 0; v < 8; ++v) {
        int m = mBase + v;
        bF[v] = bias[m];
        bI[v] = bias[m + H];
        bO[v] = bias[m + 2 * H];
        bG[v] = bias[m + 3 * H];
    }

    float* outH = out;
    float* outC = out + (size_t)H * Bn;

#pragma unroll
    for (int j = 0; j < 4; ++j) {
        int n = (nt0 + j) * 16 + nl;
        float zn  = z[n];
        float zbn = zb[n];
        float wTop = zn;
        float wCpy = (1.0f - zn) * (1.0f - zbn);
        float wUpd = (1.0f - zn) * zbn;
#pragma unroll
        for (int v = 0; v < 8; ++v) {
            int m = mBase + v;
            size_t idx = (size_t)m * Bn + n;
            float fg = sigmoidf_(acc[0][j][v] + bF[v]);
            float ig = sigmoidf_(acc[1][j][v] + bI[v]);
            float og = sigmoidf_(acc[2][j][v] + bO[v]);
            float gg = tanhf(acc[3][j][v] + bG[v]);
            float cv = c[idx];
            float hv = h[idx];
            float iag = ig * gg;
            float cNew = wTop * iag + wCpy * cv + wUpd * (fg * cv + iag);
            float tc = tanhf(cNew);
            float hNew = wTop * og * tc + wCpy * hv + wUpd * og * tc;
            outH[idx] = hNew;
            outC[idx] = cNew;
        }
    }
}

// Single ragged row 4096: z_gate -> hard-sigmoid -> binary threshold.
__global__ __launch_bounds__(256) void hmlstm_zrow(
        const float* __restrict__ Whh, const float* __restrict__ U11,
        const float* __restrict__ U21, const float* __restrict__ bias,
        const float* __restrict__ hb,  const float* __restrict__ h,
        const float* __restrict__ ht,  const float* __restrict__ z,
        float* __restrict__ out) {
    int n = blockIdx.x * blockDim.x + threadIdx.x;
    float zn = z[n];
    const float* w0 = Whh + (size_t)(4 * H) * H;
    const float* w1 = U11 + (size_t)(4 * H) * H;
    const float* w2 = U21 + (size_t)(4 * H) * H;
    float acc = bias[4 * H];
    for (int k = 0; k < H; ++k) acc += w0[k] * hb[(size_t)k * Bn + n];
    for (int k = 0; k < H; ++k) acc += w1[k] * (zn * h[(size_t)k * Bn + n]);
    for (int k = 0; k < H; ++k) acc += w2[k] * (zn * ht[(size_t)k * Bn + n]);
    float zhat = fminf(fmaxf((acc + 1.0f) * 0.5f, 0.0f), 1.0f);
    out[(size_t)2 * H * Bn + n] = (zhat > 0.5f) ? 1.0f : 0.0f;
}

extern "C" void kernel_launch(void* const* d_in, const int* in_sizes, int n_in,
                              void* d_out, int out_size, void* d_ws, size_t ws_size,
                              hipStream_t stream) {
    (void)in_sizes; (void)n_in; (void)out_size; (void)ws_size;
    const float* c_in  = (const float*)d_in[0];
    const float* hb    = (const float*)d_in[1];
    const float* h_in  = (const float*)d_in[2];
    const float* ht    = (const float*)d_in[3];
    const float* z     = (const float*)d_in[4];
    const float* zbot  = (const float*)d_in[5];
    const float* Whh   = (const float*)d_in[6];
    const float* U11   = (const float*)d_in[7];
    const float* U21   = (const float*)d_in[8];
    const float* bias  = (const float*)d_in[9];
    float* out = (float*)d_out;

    // Workspace layout (all 32B-aligned):
    //   Apack: 4 gates x 64 mTiles x 96 kTiles x 512 bf16 = 25,165,824 B
    //   Bpack: 128 nTiles x 96 kTiles x 512 bf16          = 12,582,912 B
    size_t aBytes = (size_t)4 * MT * KT * 32 * 16 * sizeof(bf16_t);
    bf16_t* Apack = (bf16_t*)d_ws;
    bf16_t* Bpack = (bf16_t*)((char*)d_ws + aBytes);

    pack_A<<<(4 * MT * KT * 32) / 256, 256, 0, stream>>>(Whh, U11, U21, Apack);
    pack_B<<<(NT * KT * 32) / 256, 256, 0, stream>>>(hb, h_in, ht, z, Bpack);
    hmlstm_zrow<<<Bn / 256, 256, 0, stream>>>(Whh, U11, U21, bias, hb, h_in, ht, z, out);
    hmlstm_gemm<<<(MT / 2) * (NT / 16), 256, 0, stream>>>(Apack, Bpack, bias, c_in,
                                                          h_in, z, zbot, out);
}
